// multi_query_atnn_20598663152351
// MI455X (gfx1250) — matrix operand order repaired and checked by reading, not compiled, not run on hardware
//
#include <hip/hip_runtime.h>

typedef unsigned short u16;
typedef unsigned int   u32;
typedef unsigned long long u64;

typedef __attribute__((ext_vector_type(16))) __bf16 v16bf;
typedef __attribute__((ext_vector_type(8)))  __bf16 v8bf;
typedef __attribute__((ext_vector_type(8)))  float  v8f;

#define DIM 2048
#define SEQ 2048
#define NH  16
#define HD  128
#define BS  2
#define MTOT (BS*SEQ)   // 4096

static __device__ __forceinline__ u16 bf16_rne(float f) {
    u32 u = __float_as_uint(f);
    u32 r = u + 0x7FFFu + ((u >> 16) & 1u);
    return (u16)(r >> 16);
}

static __device__ __forceinline__ v16bf cat8(v8bf a, v8bf b) {
    return __builtin_shufflevector(a, b, 0,1,2,3,4,5,6,7,8,9,10,11,12,13,14,15);
}

static __device__ __forceinline__ v8f wmma_bf16(v16bf a, v16bf b, v8f c) {
    return __builtin_amdgcn_wmma_f32_16x16x32_bf16(
        /*neg_a=*/false, a, /*neg_b=*/false, b,
        /*c_mod=*/(short)0, c, /*reuse_a=*/false, /*reuse_b=*/false);
}

// CDNA5 async global->LDS copy, 16 bytes per lane (ASYNCcnt-tracked).
// lds_byte: per-lane LDS byte offset; gaddr: per-lane 64-bit global address.
static __device__ __forceinline__ void async_ld16(unsigned lds_byte, const u16* gaddr) {
    asm volatile("global_load_async_to_lds_b128 %0, %1, off"
                 :: "v"(lds_byte), "v"((u64)(size_t)gaddr)
                 : "memory");
}

// ---------------------------------------------------------------- converts
__global__ __launch_bounds__(256) void cvt_bf16_kernel(const float* __restrict__ src,
                                                       u16* __restrict__ dst, long n) {
    long i = (long)blockIdx.x * blockDim.x + threadIdx.x;
    long stride = (long)gridDim.x * blockDim.x;
    for (; i < n; i += stride) dst[i] = bf16_rne(src[i]);
}

// w[K][N] (row-major fp32) -> wT[N][K] (row-major bf16)
__global__ __launch_bounds__(256) void transpose_cvt_kernel(const float* __restrict__ w,
                                                            u16* __restrict__ wT,
                                                            int K, int N) {
    long total = (long)K * N;
    long i = (long)blockIdx.x * blockDim.x + threadIdx.x;
    long stride = (long)gridDim.x * blockDim.x;
    for (; i < total; i += stride) {
        long n = i / K;
        long k = i - n * K;
        wT[i] = bf16_rne(w[k * N + n]);
    }
}

// v[b*SEQ + t][HD] bf16 -> vT[b][HD][SEQ] bf16
__global__ __launch_bounds__(256) void transpose_v_kernel(const u16* __restrict__ v,
                                                          u16* __restrict__ vT) {
    long total = (long)BS * HD * SEQ;
    long i = (long)blockIdx.x * blockDim.x + threadIdx.x;
    long stride = (long)gridDim.x * blockDim.x;
    for (; i < total; i += stride) {
        long t = i % SEQ;
        long d = (i / SEQ) % HD;
        long b = i / ((long)SEQ * HD);
        vT[i] = v[(b * SEQ + t) * HD + d];
    }
}

// ---------------------------------------------------------------- GEMM
// C[M,N] = A[M,K](bf16,row) @ BT[N,K](bf16,row; B pre-transposed) + bias.
// Block = 256 threads = 8 waves, block tile 128(M) x 128(N), K-step 32.
// Double-buffered async global->LDS staging (32 KB LDS), WMMA feeds from LDS.
// Wave (wm,wn) = (wave>>1, wave&1) computes a 32x64 sub-tile (2x4 WMMA tiles).
template <bool F32OUT>
__global__ __launch_bounds__(256) void gemm_async_kernel(const u16* __restrict__ A,
                                                         const u16* __restrict__ BT,
                                                         const float* __restrict__ bias,
                                                         void* __restrict__ Cv,
                                                         int M, int N, int K) {
    // Single shared block => LDS byte offsets are offsets into this array.
    // layout: sA[2][128][32] at 0, sB[2][128][32] at 16384 bytes.
    __shared__ u16 smem[2 * 128 * 32 + 2 * 128 * 32];

    const int tid  = threadIdx.x;
    const int wave = tid >> 5;
    const int lane = tid & 31;
    const int hi   = lane >> 4;
    const int ln   = lane & 15;
    const int wm   = wave >> 1;      // 0..3  -> M offset wm*32
    const int wn   = wave & 1;       // 0..1  -> N offset wn*64
    const int m0   = blockIdx.y * 128;
    const int n0   = blockIdx.x * 128;

    // per-thread async-copy geometry: each thread moves 2x16B of A and 2x16B of B
    const int rowL = tid >> 1;            // 0..127 (tile row for A and B)
    const int coff = (tid & 1) * 16;      // element offset in 32-elem row
    const u16* gA = A  + (size_t)(m0 + rowL) * K + coff;
    const u16* gB = BT + (size_t)(n0 + rowL) * K + coff;
    const unsigned lA = (unsigned)(rowL * 64 + coff * 2);           // bytes in A tile
    const unsigned lB = (unsigned)(16384 + rowL * 64 + coff * 2);   // bytes in B tile

    auto issue = [&](int p, int k0) {
        const unsigned pb = (unsigned)(p * 8192);
        async_ld16(pb + lA,      gA + k0);
        async_ld16(pb + lA + 16, gA + k0 + 8);
        async_ld16(pb + lB,      gB + k0);
        async_ld16(pb + lB + 16, gB + k0 + 8);
    };

    const int abase = hi ? 16 : 0;   // A frag: lanes<16 -> K 0..15; else 16..31
    const int koff  = hi ? 16 : 0;   // B frag: lanes<16 -> K 0..15; else 16..31

    v8f acc[2][4] = {};

    issue(0, 0);
    int p = 0;
    for (int k0 = 0; k0 < K; k0 += 32) {
        const int kn = (k0 + 32 < K) ? (k0 + 32) : k0;  // last iter: benign re-issue
        issue(p ^ 1, kn);
        asm volatile("s_wait_asynccnt 4" ::: "memory"); // tile for buffer p has landed
        __syncthreads();

        const u16* sA = smem + p * 4096;                // elements
        const u16* sB = smem + 8192 + p * 4096;

        v16bf af[2];
#pragma unroll
        for (int f = 0; f < 2; f++) {
            const u16* ar = sA + (wm * 32 + f * 16 + ln) * 32 + abase;
            af[f] = cat8(*reinterpret_cast<const v8bf*>(ar),
                         *reinterpret_cast<const v8bf*>(ar + 8));
        }
#pragma unroll
        for (int j = 0; j < 4; j++) {
            const v16bf bfr = *reinterpret_cast<const v16bf*>(
                sB + (wn * 64 + j * 16 + ln) * 32 + koff);
#pragma unroll
            for (int f = 0; f < 2; f++)
                acc[f][j] = wmma_bf16(af[f], bfr, acc[f][j]);
        }
        __syncthreads();   // everyone done reading buffer p before it is overwritten
        p ^= 1;
    }

#pragma unroll
    for (int f = 0; f < 2; f++)
#pragma unroll
        for (int j = 0; j < 4; j++) {
            const int col = n0 + wn * 64 + j * 16 + ln;
            const float bv = bias[col];
#pragma unroll
            for (int r = 0; r < 8; r++) {
                const int row = m0 + wm * 32 + f * 16 + r + 8 * hi;
                const float val = acc[f][j][r] + bv;
                if (F32OUT)
                    ((float*)Cv)[(size_t)row * N + col] = val;
                else
                    ((u16*)Cv)[(size_t)row * N + col] = bf16_rne(val);
            }
        }
}

// ---------------------------------------------------------------- attention
// grid: (SEQ/64, NH, BS); block 128 = 4 waves; wave handles 16 query rows.
__global__ __launch_bounds__(128) void attn_kernel(const u16* __restrict__ qb,
                                                   const u16* __restrict__ kb,
                                                   const u16* __restrict__ vTb,
                                                   u16* __restrict__ ob) {
    const int wave = threadIdx.x >> 5;
    const int lane = threadIdx.x & 31;
    const int hi   = lane >> 4;
    const int ln   = lane & 15;
    const int b = blockIdx.z;
    const int h = blockIdx.y;
    const int s_base = blockIdx.x * 64 + wave * 16;

    __shared__ u16 ldsP[4][16 * 32];
    u16* myP = ldsP[wave];

    const float scale = 0.08838834764831845f;  // 1/sqrt(128)

    // Q fragments for full head dim (4 x K=32)
    const u16* qrow  = qb + (size_t)(b * SEQ + s_base + ln) * DIM + h * HD;
    const int  abase = hi ? 16 : 0;
    const int  koff  = hi ? 16 : 0;
    v16bf aq[4];
#pragma unroll
    for (int kk = 0; kk < 4; kk++) {
        v8bf lo = *reinterpret_cast<const v8bf*>(qrow + kk * 32 + abase);
        v8bf hh = *reinterpret_cast<const v8bf*>(qrow + kk * 32 + abase + 8);
        aq[kk] = cat8(lo, hh);
    }

    float mrun[8], lrun[8];
    v8f oacc[8] = {};
#pragma unroll
    for (int r = 0; r < 8; r++) { mrun[r] = -1e30f; lrun[r] = 0.0f; }

    const u16* kbase = kb  + (size_t)b * SEQ * HD;
    const u16* vtb   = vTb + (size_t)b * HD * SEQ;

    for (int t0 = 0; t0 < SEQ; t0 += 32) {
        // ---- scores: S[16 x 32] over head dim (4 WMMA K-steps, 2 N-subtiles)
        v8f s0 = {}, s1 = {};
#pragma unroll
        for (int kk = 0; kk < 4; kk++) {
            const v16bf b0 = *reinterpret_cast<const v16bf*>(
                kbase + (size_t)(t0 + ln) * HD + kk * 32 + koff);
            const v16bf b1 = *reinterpret_cast<const v16bf*>(
                kbase + (size_t)(t0 + 16 + ln) * HD + kk * 32 + koff);
            s0 = wmma_bf16(aq[kk], b0, s0);
            s1 = wmma_bf16(aq[kk], b1, s1);
        }

        // ---- online softmax (row = r + 8*hi; 16 lanes of a half hold the row)
#pragma unroll
        for (int r = 0; r < 8; r++) {
            float a0 = s0[r] * scale;
            float a1 = s1[r] * scale;
            float vmax = fmaxf(a0, a1);
            vmax = fmaxf(vmax, __shfl_xor(vmax, 1, 32));
            vmax = fmaxf(vmax, __shfl_xor(vmax, 2, 32));
            vmax = fmaxf(vmax, __shfl_xor(vmax, 4, 32));
            vmax = fmaxf(vmax, __shfl_xor(vmax, 8, 32));
            float mnew = fmaxf(mrun[r], vmax);
            float p0 = __expf(a0 - mnew);
            float p1 = __expf(a1 - mnew);
            float psum = p0 + p1;
            psum += __shfl_xor(psum, 1, 32);
            psum += __shfl_xor(psum, 2, 32);
            psum += __shfl_xor(psum, 4, 32);
            psum += __shfl_xor(psum, 8, 32);
            float alpha = __expf(mrun[r] - mnew);
            lrun[r] = lrun[r] * alpha + psum;
            mrun[r] = mnew;
#pragma unroll
            for (int j = 0; j < 8; j++) oacc[j][r] *= alpha;
            const int row = r + 8 * hi;
            myP[row * 32 + ln]      = bf16_rne(p0);
            myP[row * 32 + 16 + ln] = bf16_rne(p1);
        }
        __syncthreads();

        // ---- re-layout P (D-format -> A-format) via LDS
        const u16* prow = myP + ln * 32 + abase;
        v8bf plo = *reinterpret_cast<const v8bf*>(prow);
        v8bf phi = *reinterpret_cast<const v8bf*>(prow + 8);
        v16bf pfrag = cat8(plo, phi);

        // ---- O += P @ V  (8 d-subtiles of 16)
#pragma unroll
        for (int j = 0; j < 8; j++) {
            const v16bf vb = *reinterpret_cast<const v16bf*>(
                vtb + (size_t)(j * 16 + ln) * SEQ + t0 + koff);
            oacc[j] = wmma_bf16(pfrag, vb, oacc[j]);
        }
        __syncthreads();
    }

    // ---- normalize + store bf16 O tile
#pragma unroll
    for (int j = 0; j < 8; j++) {
        const int col = h * HD + j * 16 + ln;
#pragma unroll
        for (int r = 0; r < 8; r++) {
            const int s = s_base + r + 8 * hi;
            float val = oacc[j][r] / lrun[r];
            ob[(size_t)(b * SEQ + s) * DIM + col] = bf16_rne(val);
        }
    }
}

// ---------------------------------------------------------------- launch
extern "C" void kernel_launch(void* const* d_in, const int* in_sizes, int n_in,
                              void* d_out, int out_size, void* d_ws, size_t ws_size,
                              hipStream_t stream) {
    const float* x  = (const float*)d_in[0];
    const float* wq = (const float*)d_in[1];
    const float* bq = (const float*)d_in[2];
    const float* wk = (const float*)d_in[3];
    const float* bk = (const float*)d_in[4];
    const float* wv = (const float*)d_in[5];
    const float* bv = (const float*)d_in[6];
    const float* wo = (const float*)d_in[7];
    const float* bo = (const float*)d_in[8];
    float* out = (float*)d_out;

    char* p = (char*)d_ws;
    u16* xb  = (u16*)p; p += (size_t)MTOT * DIM * 2;        // x bf16
    u16* wqT = (u16*)p; p += (size_t)DIM * DIM * 2;         // wq^T bf16 [2048][2048]
    u16* wkT = (u16*)p; p += (size_t)HD  * DIM * 2;         // wk^T bf16 [128][2048]
    u16* wvT = (u16*)p; p += (size_t)HD  * DIM * 2;         // wv^T bf16
    u16* woT = (u16*)p; p += (size_t)DIM * DIM * 2;         // wo^T bf16
    u16* qB  = (u16*)p; p += (size_t)MTOT * DIM * 2;        // q bf16 [4096][2048]
    u16* kB  = (u16*)p; p += (size_t)MTOT * HD * 2;         // k bf16 [4096][128]
    u16* vB  = (u16*)p; p += (size_t)MTOT * HD * 2;         // v bf16 [4096][128]
    u16* vT  = (u16*)p; p += (size_t)BS * HD * SEQ * 2;     // v^T bf16 [2][128][2048]
    u16* oB  = (u16*)p; p += (size_t)MTOT * DIM * 2;        // o bf16 [4096][2048]

    // 1) conversions
    cvt_bf16_kernel<<<4096, 256, 0, stream>>>(x, xb, (long)MTOT * DIM);
    transpose_cvt_kernel<<<4096, 256, 0, stream>>>(wq, wqT, DIM, DIM);
    transpose_cvt_kernel<<<1024, 256, 0, stream>>>(wk, wkT, DIM, HD);
    transpose_cvt_kernel<<<1024, 256, 0, stream>>>(wv, wvT, DIM, HD);
    transpose_cvt_kernel<<<4096, 256, 0, stream>>>(wo, woT, DIM, DIM);

    // 2) projections (bf16 out), async-LDS double-buffered WMMA GEMM
    gemm_async_kernel<false><<<dim3(DIM / 128, MTOT / 128), 256, 0, stream>>>(
        xb, wqT, bq, qB, MTOT, DIM, DIM);
    gemm_async_kernel<false><<<dim3(HD / 128, MTOT / 128), 256, 0, stream>>>(
        xb, wkT, bk, kB, MTOT, HD, DIM);
    gemm_async_kernel<false><<<dim3(HD / 128, MTOT / 128), 256, 0, stream>>>(
        xb, wvT, bv, vB, MTOT, HD, DIM);

    // 3) transpose V for contiguous PV B-fragments
    transpose_v_kernel<<<1024, 256, 0, stream>>>(vB, vT);

    // 4) flash attention
    attn_kernel<<<dim3(SEQ / 64, NH, BS), 128, 0, stream>>>(qB, kB, vT, oB);

    // 5) output projection (fp32 out)
    gemm_async_kernel<true><<<dim3(DIM / 128, MTOT / 128), 256, 0, stream>>>(
        oB, woT, bo, out, MTOT, DIM, DIM);
}
